// NetGATFix_69569880261288
// MI455X (gfx1250) — matrix operand order repaired and checked by reading, not compiled, not run on hardware
//
#include <hip/hip_runtime.h>
#include <hip/hip_bf16.h>

// ---------------------------------------------------------------------------
// NetGATFix for MI455X (gfx1250, wave32).
// Dense projections use v_wmma_f32_16x16x32_f16 with fully-static fragment
// loads (global_load_b128); edge softmax/scatter uses L2-resident gathers +
// f32 atomics with the head-mean folded into the scatter.
// ---------------------------------------------------------------------------

#define NNODES 100000
#define NEDGES 3200000
#define NHEADS 6

typedef __attribute__((ext_vector_type(16))) _Float16 v16h;
typedef __attribute__((ext_vector_type(8)))  float    v8f;

// ---- monotone float<->uint encoding for atomicMax on floats ----------------
__device__ __forceinline__ unsigned f2o(float f) {
    unsigned u = __float_as_uint(f);
    return (u & 0x80000000u) ? ~u : (u | 0x80000000u);
}
__device__ __forceinline__ float o2f(unsigned u) {
    return __uint_as_float((u & 0x80000000u) ? (u & 0x7fffffffu) : ~u);
}

__device__ __forceinline__ float selu_f(float v) {
    const float kScale = 1.0507009873554805f;
    const float kAlpha = 1.6732632423543772f;
    return kScale * (v > 0.0f ? v : kAlpha * (__expf(v) - 1.0f));
}

__device__ __forceinline__ float lrelu02(float v) {
    return v > 0.0f ? v : 0.2f * v;
}

// ---------------------------------------------------------------------------
// K0: h0 = fix(selu(x @ W0^T + b0)).
// ---------------------------------------------------------------------------
__global__ void k_input(const float* __restrict__ x, const float* __restrict__ W0,
                        const float* __restrict__ b0, float* __restrict__ h) {
    int tid = blockIdx.x * blockDim.x + threadIdx.x;
    if (tid >= NNODES * 16) return;
    int n = tid >> 4, o = tid & 15;
    const float* xr = x + (size_t)n * 16;
    const float* wr = W0 + (size_t)o * 16;
    float v = b0[o];
    #pragma unroll
    for (int k = 0; k < 16; ++k) v += xr[k] * wr[k];
    v = selu_f(v);
    if (o == 0) { float x0 = xr[0]; if (x0 == 0.0f) v = 0.0f; else if (x0 == 1.0f) v = 1.0f; }
    if (o == 1) { float x1 = xr[1]; if (x1 == 1.0f) v = 1.0f; else if (x1 == 0.0f) v = 0.0f; }
    h[tid] = v;
}

// ---------------------------------------------------------------------------
// K1: hproj[N, HF] = h[N, Fin] @ W[HF, Fin]^T via V_WMMA_F32_16X16X32_F16.
// Templated on <Fin, HF> so every fragment load is a contiguous float4
// (global_load_b128) and all K/HF guards are compile-time or branchless.
//
// A-fragment (16-bit A 16x32, ISA 7.12.2): lane = khi*16 + m,
//   a[0..7]  -> K in [8*khi, 8*khi+8)
//   a[8..15] -> K in [16+8*khi, 24+8*khi)        (zero-padded when Fin==16)
// B-fragment (32x16): col = lane&15, b[0..7] -> K = 8*khi + j, b[8..15] -> K = 16+8*khi + (j-8).
// D (f32 16x16): row = 8*khi + r, col = lane&15.
// ---------------------------------------------------------------------------
template <int Fin, int HF>
__global__ __launch_bounds__(32) void k_proj_wmma(const float* __restrict__ h,
                                                  const float* __restrict__ W,
                                                  float* __restrict__ hproj) {
    const int lane = threadIdx.x;      // 0..31
    const int m    = lane & 15;
    const int khi  = lane >> 4;
    const int row0 = blockIdx.x * 16;  // N == 6250*16 exactly, no M guard

    // ---- A fragment: two (or four) b128 loads, converted to f16 -----------
    const float4* arow = (const float4*)(h + (size_t)(row0 + m) * Fin);
    float4 a0 = arow[2 * khi];
    float4 a1 = arow[2 * khi + 1];
    float4 a2 = make_float4(0.f, 0.f, 0.f, 0.f);
    float4 a3 = make_float4(0.f, 0.f, 0.f, 0.f);
    if (Fin == 32) { a2 = arow[4 + 2 * khi]; a3 = arow[5 + 2 * khi]; }

    v16h a;
    a[0]  = (_Float16)a0.x; a[1]  = (_Float16)a0.y; a[2]  = (_Float16)a0.z; a[3]  = (_Float16)a0.w;
    a[4]  = (_Float16)a1.x; a[5]  = (_Float16)a1.y; a[6]  = (_Float16)a1.z; a[7]  = (_Float16)a1.w;
    a[8]  = (_Float16)a2.x; a[9]  = (_Float16)a2.y; a[10] = (_Float16)a2.z; a[11] = (_Float16)a2.w;
    a[12] = (_Float16)a3.x; a[13] = (_Float16)a3.y; a[14] = (_Float16)a3.z; a[15] = (_Float16)a3.w;

    constexpr int ntiles = (HF + 15) / 16;
    #pragma unroll
    for (int t = 0; t < ntiles; ++t) {
        const int  col    = t * 16 + m;                       // output feature
        const bool cvalid = ((HF & 15) == 0) || (col < HF);   // tail tile only
        const int  colc   = cvalid ? col : 0;                 // safe address

        // ---- B fragment: four b128 loads, branchless zero for pads --------
        const float* wrow = W + (size_t)colc * Fin;
        float wv[16];
        if (Fin == 32) {
            const float4* wr = (const float4*)(wrow + 8 * khi);
            float4 q0 = wr[0], q1 = wr[1], q2 = wr[4], q3 = wr[5];
            wv[0]=q0.x; wv[1]=q0.y; wv[2]=q0.z; wv[3]=q0.w;
            wv[4]=q1.x; wv[5]=q1.y; wv[6]=q1.z; wv[7]=q1.w;
            wv[8]=q2.x; wv[9]=q2.y; wv[10]=q2.z; wv[11]=q2.w;
            wv[12]=q3.x; wv[13]=q3.y; wv[14]=q3.z; wv[15]=q3.w;
        } else {
            // b[0..7] -> K = 8*khi + j; b[8..15] multiply zero A pads.
            const float4* wr = (const float4*)(wrow + 8 * khi);
            float4 q0 = wr[0], q1 = wr[1], q2 = q0, q3 = q1;
            wv[0]=q0.x; wv[1]=q0.y; wv[2]=q0.z; wv[3]=q0.w;
            wv[4]=q1.x; wv[5]=q1.y; wv[6]=q1.z; wv[7]=q1.w;
            wv[8]=q2.x; wv[9]=q2.y; wv[10]=q2.z; wv[11]=q2.w;
            wv[12]=q3.x; wv[13]=q3.y; wv[14]=q3.z; wv[15]=q3.w;
        }
        float scale = cvalid ? 1.0f : 0.0f;


        v16h b;
        #pragma unroll
        for (int j = 0; j < 16; ++j) b[j] = (_Float16)(wv[j] * scale);

        v8f c = {};
        c = __builtin_amdgcn_wmma_f32_16x16x32_f16(
                /*neg_a=*/false, a, /*neg_b=*/false, b,
                /*c_mod=*/(short)0, c, /*reuse_a=*/false, /*reuse_b=*/false);

        #pragma unroll
        for (int r = 0; r < 8; ++r) {
            int row = row0 + khi * 8 + r;
            if (cvalid) hproj[(size_t)row * HF + col] = c[r];
        }
    }
}

// ---------------------------------------------------------------------------
// K2: per (node, head) attention logits + init of softmax state.
// ---------------------------------------------------------------------------
template <int F>
__global__ void k_al_init(const float* __restrict__ hproj,
                          const float* __restrict__ a_s, const float* __restrict__ a_d,
                          float* __restrict__ als, float* __restrict__ ald,
                          unsigned* __restrict__ menc, float* __restrict__ ssum) {
    constexpr int HF = NHEADS * F;
    int tid = blockIdx.x * blockDim.x + threadIdx.x;  // N*H
    if (tid >= NNODES * NHEADS) return;
    int n = tid / NHEADS, hh = tid % NHEADS;
    const float* hp = hproj + (size_t)n * HF + hh * F;
    const float* as = a_s + hh * F;
    const float* ad = a_d + hh * F;
    float s = 0.0f, d = 0.0f;
    #pragma unroll
    for (int f = 0; f < F; ++f) { float v = hp[f]; s += v * as[f]; d += v * ad[f]; }
    als[tid] = s; ald[tid] = d;
    menc[tid] = 0u;        // encoded value below every real float
    ssum[tid] = 0.0f;
}

__global__ void k_zero(float* __restrict__ p, int n) {
    int tid = blockIdx.x * blockDim.x + threadIdx.x;
    if (tid < n) p[tid] = 0.0f;
}

// ---------------------------------------------------------------------------
// K3: segment max of e = leaky_relu(al_s[src]+al_d[dst]) over dst, per head.
// ---------------------------------------------------------------------------
__global__ void k_edge_max(const int* __restrict__ ei,
                           const float* __restrict__ als, const float* __restrict__ ald,
                           unsigned* __restrict__ menc) {
    int e = blockIdx.x * blockDim.x + threadIdx.x;
    if (e >= NEDGES + NNODES) return;
    int src, dst;
    if (e < NEDGES) { src = ei[e]; dst = ei[NEDGES + e]; } else { src = dst = e - NEDGES; }
    #pragma unroll
    for (int hh = 0; hh < NHEADS; ++hh) {
        float lg = lrelu02(als[src * NHEADS + hh] + ald[dst * NHEADS + hh]);
        atomicMax(&menc[dst * NHEADS + hh], f2o(lg));
    }
}

// ---------------------------------------------------------------------------
// K4: segment sum of exp(e - m[dst]) over dst, per head.
// ---------------------------------------------------------------------------
__global__ void k_edge_sum(const int* __restrict__ ei,
                           const float* __restrict__ als, const float* __restrict__ ald,
                           const unsigned* __restrict__ menc, float* __restrict__ ssum) {
    int e = blockIdx.x * blockDim.x + threadIdx.x;
    if (e >= NEDGES + NNODES) return;
    int src, dst;
    if (e < NEDGES) { src = ei[e]; dst = ei[NEDGES + e]; } else { src = dst = e - NEDGES; }
    #pragma unroll
    for (int hh = 0; hh < NHEADS; ++hh) {
        float lg = lrelu02(als[src * NHEADS + hh] + ald[dst * NHEADS + hh]);
        float p  = __expf(lg - o2f(menc[dst * NHEADS + hh]));
        atomicAdd(&ssum[dst * NHEADS + hh], p);
    }
}

// ---------------------------------------------------------------------------
// K5: scatter.  One thread per (edge, f); head-mean folded in -> single
// atomicAdd per (edge,f) into acc[N,F].  F constexpr => %/ are shifts.
// ---------------------------------------------------------------------------
template <int F>
__global__ void k_edge_scatter(const int* __restrict__ ei,
                               const float* __restrict__ als, const float* __restrict__ ald,
                               const unsigned* __restrict__ menc, const float* __restrict__ ssum,
                               const float* __restrict__ hproj, float* __restrict__ acc) {
    constexpr int HF = NHEADS * F;
    long long tid = (long long)blockIdx.x * blockDim.x + threadIdx.x;
    long long total = (long long)(NEDGES + NNODES) * F;
    if (tid >= total) return;
    int f = (int)(tid % F);
    long long e = tid / F;
    int src, dst;
    if (e < NEDGES) { src = ei[e]; dst = ei[NEDGES + e]; } else { src = dst = (int)(e - NEDGES); }

    const float* hp = hproj + (size_t)src * HF + f;
    __builtin_prefetch(hp, 0, 1);   // gfx1250 global_prefetch_b8

    float val = 0.0f;
    #pragma unroll
    for (int hh = 0; hh < NHEADS; ++hh) {
        float lg = lrelu02(als[src * NHEADS + hh] + ald[dst * NHEADS + hh]);
        float alpha = __expf(lg - o2f(menc[dst * NHEADS + hh])) / ssum[dst * NHEADS + hh];
        val += hp[hh * F] * alpha;
    }
    atomicAdd(&acc[(size_t)dst * F + f], val * (1.0f / NHEADS));
}

// ---------------------------------------------------------------------------
// K6: out = fix(selu(acc + b)) per (node, f).
// ---------------------------------------------------------------------------
template <int F>
__global__ void k_finalize(const float* __restrict__ acc, const float* __restrict__ bias,
                           const float* __restrict__ x, float* __restrict__ out) {
    int tid = blockIdx.x * blockDim.x + threadIdx.x;
    if (tid >= NNODES * F) return;
    int n = tid / F, f = tid % F;
    float v = selu_f(acc[tid] + bias[f]);
    if (f == 0) { float x0 = x[(size_t)n * 16 + 0]; if (x0 == 0.0f) v = 0.0f; else if (x0 == 1.0f) v = 1.0f; }
    if (f == 1) { float x1 = x[(size_t)n * 16 + 1]; if (x1 == 1.0f) v = 1.0f; else if (x1 == 0.0f) v = 0.0f; }
    out[tid] = v;
}

// ---------------------------------------------------------------------------
struct Scratch {
    float*    hbuf;
    float*    hproj;
    float*    als;
    float*    ald;
    unsigned* menc;
    float*    ssum;
    float*    acc;
};

template <int Fin, int F>
static void run_layer(const float* W, const float* a_s, const float* a_d,
                      const float* bias, const float* x, const int* ei,
                      const Scratch& S, float* out, hipStream_t stream) {
    constexpr int HF = NHEADS * F;
    const int B = 256;

    k_proj_wmma<Fin, HF><<<dim3(NNODES / 16), 32, 0, stream>>>(S.hbuf, W, S.hproj);

    k_al_init<F><<<(NNODES * NHEADS + B - 1) / B, B, 0, stream>>>(
        S.hproj, a_s, a_d, S.als, S.ald, S.menc, S.ssum);

    k_zero<<<(NNODES * F + B - 1) / B, B, 0, stream>>>(S.acc, NNODES * F);

    const int ne = NEDGES + NNODES;
    k_edge_max<<<(ne + B - 1) / B, B, 0, stream>>>(ei, S.als, S.ald, S.menc);
    k_edge_sum<<<(ne + B - 1) / B, B, 0, stream>>>(ei, S.als, S.ald, S.menc, S.ssum);

    long long tot = (long long)ne * F;
    k_edge_scatter<F><<<(unsigned)((tot + B - 1) / B), B, 0, stream>>>(
        ei, S.als, S.ald, S.menc, S.ssum, S.hproj, S.acc);

    k_finalize<F><<<(NNODES * F + B - 1) / B, B, 0, stream>>>(S.acc, bias, x, out);
}

extern "C" void kernel_launch(void* const* d_in, const int* in_sizes, int n_in,
                              void* d_out, int out_size, void* d_ws, size_t ws_size,
                              hipStream_t stream) {
    const float* x  = (const float*)d_in[0];
    const int*   ei = (const int*)d_in[1];
    const float* W0 = (const float*)d_in[2];
    const float* b0 = (const float*)d_in[3];

    // Workspace layout (floats): ~112 MB total.
    Scratch S;
    S.hbuf  = (float*)d_ws;                                  // N*32 (reused)
    S.hproj = S.hbuf  + (size_t)NNODES * 32;                 // N*192
    S.als   = S.hproj + (size_t)NNODES * 192;                // N*H
    S.ald   = S.als   + (size_t)NNODES * NHEADS;             // N*H
    S.menc  = (unsigned*)(S.ald + (size_t)NNODES * NHEADS);  // N*H
    S.ssum  = (float*)(S.menc + (size_t)NNODES * NHEADS);    // N*H
    S.acc   = S.ssum  + (size_t)NNODES * NHEADS;             // N*32

    const int B = 256;
    k_input<<<(NNODES * 16 + B - 1) / B, B, 0, stream>>>(x, W0, b0, S.hbuf);

    run_layer<16, 32>((const float*)d_in[4],  (const float*)d_in[5],
                      (const float*)d_in[6],  (const float*)d_in[7],
                      x, ei, S, S.hbuf, stream);
    run_layer<32, 32>((const float*)d_in[8],  (const float*)d_in[9],
                      (const float*)d_in[10], (const float*)d_in[11],
                      x, ei, S, S.hbuf, stream);
    run_layer<32, 16>((const float*)d_in[12], (const float*)d_in[13],
                      (const float*)d_in[14], (const float*)d_in[15],
                      x, ei, S, S.hbuf, stream);
    run_layer<16, 2>((const float*)d_in[16], (const float*)d_in[17],
                     (const float*)d_in[18], (const float*)d_in[19],
                     x, ei, S, (float*)d_out, stream);
}
